// BetaVAEMark7_34376918237630
// MI455X (gfx1250) — matrix operand order repaired and checked by reading, not compiled, not run on hardware
//
#include <hip/hip_runtime.h>
#include <hip/hip_bf16.h>

typedef __attribute__((ext_vector_type(16))) __bf16 v16bf;
typedef __attribute__((ext_vector_type(8)))  float  v8f;
typedef __attribute__((ext_vector_type(4)))  int    v4i;

// ---------------------------------------------------------------------------
// Weight packing: lay out bf16 B-fragments for v_wmma_f32_16x16x32_bf16.
// B (32x16, K x N) element (k,n):  lane = n + 16*(k/16), half h = k%16.
// Packed layout: P[((tn*nKC + kc)*32 + lane)*16 + h]
// k ordering matches A-gather: k = (ci*KH + kh)*KW + kw
// tflip=1: ConvT weights (Cin,Cout,KH,KW) -> flipped+transposed OIHW
// ---------------------------------------------------------------------------
__global__ void pack_weights(const float* __restrict__ W, __bf16* __restrict__ P,
                             int Cout, int Cin, int KH, int KW,
                             int nKC, int nTN, int tflip, int total)
{
    int idx = blockIdx.x * 256 + threadIdx.x;
    if (idx >= total) return;
    int h    = idx & 15;
    int lane = (idx >> 4) & 31;
    int kc   = (idx >> 9) % nKC;
    int tn   = (idx >> 9) / nKC;
    int klocal = h + 16 * (lane >> 4);
    int nlocal = lane & 15;
    int k = kc * 32 + klocal;
    int n = tn * 16 + nlocal;
    int K = Cin * KH * KW;
    float v = 0.0f;
    if (k < K && n < Cout) {
        int ci = k / (KH * KW);
        int r  = k % (KH * KW);
        int kh = r / KW;
        int kw = r % KW;
        if (!tflip) v = W[((n * Cin + ci) * KH + kh) * KW + kw];
        else        v = W[((ci * Cout + n) * KH + (KH - 1 - kh)) * KW + (KW - 1 - kw)];
    }
    P[idx] = (__bf16)v;
}

// ---------------------------------------------------------------------------
// Generic implicit-GEMM conv / convT / dense via WMMA bf16.
// B-fragments are staged once per block into LDS via GLOBAL_LOAD_ASYNC_TO_LDS
// (ASYNCcnt), then all waves feed WMMA from ds loads.
// ih_d = oh*SH - PH + kh, valid iff ih_d>=0, ih_d%LDH==0, ih_d/LDH<IH.
// Normal conv: LDH=LDW=1. ConvT: SH=SW=1, PH=KH-1-pad, LDH=orig stride,
// weights pre flipped+transposed. Dense: KH=KW=IH=IW=OH=OW=1.
// Block = 256 threads = 8 waves; each wave owns 16 rows x (NTN*16) cols.
// M totals are multiples of 128, so EXEC is all-ones at every WMMA.
// ---------------------------------------------------------------------------
template<int CIN,int COUT,int KH,int KW,int SH,int SW,int PH,int PW,
         int LDH,int LDW,int IH,int IW,int OH,int OW,bool RELU>
__global__ __launch_bounds__(256) void conv_wmma(
    const float* __restrict__ In, const __bf16* __restrict__ Wp,
    const float* __restrict__ bias, float* __restrict__ Out)
{
    constexpr int K   = CIN * KH * KW;
    constexpr int NKC = (K + 31) / 32;
    constexpr int NTN = (COUT + 15) / 16;
    constexpr int LDS_SHORTS = NTN * NKC * 512;   // packed B fragments (bf16)
    constexpr int NCHUNK     = LDS_SHORTS / 8;    // 16-byte transfers

    __shared__ __align__(32) short ldsW[LDS_SHORTS];

    // ---- stage packed weights to LDS (async DMA path when available) ----
    {
        const int tid = threadIdx.x;
#if __has_builtin(__builtin_amdgcn_global_load_async_to_lds_b128)
        auto gsrc = (__attribute__((address_space(1))) v4i*)const_cast<__bf16*>(Wp);
        auto ldst = (__attribute__((address_space(3))) v4i*)ldsW;
        for (int i = tid; i < NCHUNK; i += 256)
            __builtin_amdgcn_global_load_async_to_lds_b128(gsrc + i, ldst + i, 0, 0);
#if __has_builtin(__builtin_amdgcn_s_wait_asynccnt)
        __builtin_amdgcn_s_wait_asynccnt(0);
#else
        asm volatile("s_wait_asynccnt 0" ::: "memory");
#endif
#else
        for (int i = tid; i < NCHUNK; i += 256)
            ((int4*)ldsW)[i] = ((const int4*)Wp)[i];
#endif
        __syncthreads();
    }

    const int lane  = threadIdx.x & 31;
    const int wave  = threadIdx.x >> 5;
    const int tileM = (blockIdx.x * 8 + wave) * 16;
    const int s     = lane >> 4;          // which K-half this lane holds for A
    const int arow  = lane & 15;          // A row for this lane

    // A-row coordinates (each lane gathers row tileM+arow)
    const int aM   = tileM + arow;
    const int aow  = aM % OW;
    const int aoh  = (aM / OW) % OH;
    const int aimg = aM / (OW * OH);
    const float* __restrict__ Ain = In + (size_t)aimg * CIN * IH * IW;

    v8f acc[NTN];
#pragma unroll
    for (int tn = 0; tn < NTN; ++tn) acc[tn] = (v8f){};

#pragma unroll
    for (int kc = 0; kc < NKC; ++kc) {
        // ---- build A fragment (16 bf16 per lane); k fully constant-folds ----
        v16bf a;
#pragma unroll
        for (int h = 0; h < 16; ++h) {
            const int klocal = h + 16 * s;
            const int k = kc * 32 + klocal;
            float v = 0.0f;
            if (k < K) {
                const int ci = k / (KH * KW);
                const int r  = k % (KH * KW);
                const int kh = r / KW;
                const int kw = r % KW;
                const int ihd = aoh * SH - PH + kh;
                const int iwd = aow * SW - PW + kw;
                bool ok = (ihd >= 0) && (iwd >= 0) &&
                          (ihd % LDH == 0) && (iwd % LDW == 0);
                const int ih = ihd / LDH;
                const int iw = iwd / LDW;
                ok = ok && (ih < IH) && (iw < IW);
                if (ok) v = Ain[(ci * IH + ih) * IW + iw];
            }
            a[h] = (__bf16)v;
        }
        // ---- one WMMA per N-tile, B fragment straight from LDS ----
#pragma unroll
        for (int tn = 0; tn < NTN; ++tn) {
            v16bf b = *(const v16bf*)&ldsW[((tn * NKC + kc) * 32 + lane) * 16];
            acc[tn] = __builtin_amdgcn_wmma_f32_16x16x32_bf16(
                false, a, false, b, (short)0, acc[tn], false, false);
        }
    }

    // ---- epilogue: C layout lane = n + 16*(m/8); vgpr r -> m = r + 8*s ----
#pragma unroll
    for (int tn = 0; tn < NTN; ++tn) {
        const int col = tn * 16 + (lane & 15);
        const float bv = bias[col < COUT ? col : 0];
#pragma unroll
        for (int r = 0; r < 8; ++r) {
            const int m   = r + 8 * s;
            const int row = tileM + m;
            const int ow2 = row % OW;
            const int oh2 = (row / OW) % OH;
            const int ni  = row / (OW * OH);
            float v = acc[tn][r] + bv;
            if (RELU) v = (v > 0.0f) ? v : 0.01f * v;
            if (col < COUT)
                Out[((ni * COUT + col) * OH + oh2) * OW + ow2] = v;
        }
    }
}

// ---------------------------------------------------------------------------
// Reparameterization: mu, logvar=clip(lv,-5,0), z = mu + eps*exp(0.5*logvar)
// ---------------------------------------------------------------------------
__global__ void vae_latent(const float* __restrict__ mu_raw,
                           const float* __restrict__ lv_raw,
                           const float* __restrict__ eps,
                           float* __restrict__ out_mu,
                           float* __restrict__ out_lv,
                           float* __restrict__ out_z, int n)
{
    int i = blockIdx.x * 256 + threadIdx.x;
    if (i >= n) return;
    float mu = mu_raw[i];
    float lv = lv_raw[i];
    lv = fminf(fmaxf(lv, -5.0f), 0.0f);
    float z = mu + eps[i] * expf(0.5f * lv);
    out_mu[i] = mu;
    out_lv[i] = lv;
    out_z[i]  = z;
}

// ---------------------------------------------------------------------------
static inline size_t packElems(int Cout, int Cin, int KH, int KW) {
    int K = Cin * KH * KW;
    int nkc = (K + 31) / 32;
    int ntn = (Cout + 15) / 16;
    return (size_t)nkc * ntn * 512;   // bf16 elements
}

extern "C" void kernel_launch(void* const* d_in, const int* in_sizes, int n_in,
                              void* d_out, int out_size, void* d_ws, size_t ws_size,
                              hipStream_t stream)
{
    (void)in_sizes; (void)n_in; (void)out_size; (void)ws_size;
    const float* x    = (const float*)d_in[0];
    const float* eps  = (const float*)d_in[1];
    const float* w1   = (const float*)d_in[2];  const float* b1   = (const float*)d_in[3];
    const float* wp1  = (const float*)d_in[4];  const float* bp1  = (const float*)d_in[5];
    const float* w2   = (const float*)d_in[6];  const float* b2   = (const float*)d_in[7];
    const float* wp2  = (const float*)d_in[8];  const float* bp2  = (const float*)d_in[9];
    const float* w3   = (const float*)d_in[10]; const float* b3   = (const float*)d_in[11];
    const float* wp3  = (const float*)d_in[12]; const float* bp3  = (const float*)d_in[13];
    const float* mu_w = (const float*)d_in[14]; const float* mu_b = (const float*)d_in[15];
    const float* lv_w = (const float*)d_in[16]; const float* lv_b = (const float*)d_in[17];
    const float* dc_w = (const float*)d_in[18]; const float* dc_b = (const float*)d_in[19];
    const float* tu1w = (const float*)d_in[20]; const float* tu1b = (const float*)d_in[21];
    const float* t1w  = (const float*)d_in[22]; const float* t1b  = (const float*)d_in[23];
    const float* tu2w = (const float*)d_in[24]; const float* tu2b = (const float*)d_in[25];
    const float* t2w  = (const float*)d_in[26]; const float* t2b  = (const float*)d_in[27];
    const float* tu3w = (const float*)d_in[28]; const float* tu3b = (const float*)d_in[29];
    const float* t3w  = (const float*)d_in[30]; const float* t3b  = (const float*)d_in[31];

    const int B = 2048;
    // workspace layout: two ping-pong activation buffers + packed weights + latents
    const size_t BUF = 209715200;               // 2048*8*50*64*4 bytes (largest layer)
    char*  ws   = (char*)d_ws;
    float* bufA = (float*)(ws);
    float* bufB = (float*)(ws + BUF);
    size_t off  = 2 * BUF;
    auto carve_bf16 = [&](size_t elems) -> __bf16* {
        off = (off + 255) & ~size_t(255);
        __bf16* p = (__bf16*)(ws + off);
        off += elems * sizeof(__bf16);
        return p;
    };
    auto carve_f32 = [&](size_t elems) -> float* {
        off = (off + 255) & ~size_t(255);
        float* p = (float*)(ws + off);
        off += elems * sizeof(float);
        return p;
    };

    struct PackDesc { const float* W; int Cout, Cin, KH, KW, tflip; __bf16* P; };
    PackDesc pk[15] = {
        { w1,   8,   6,   3, 3, 0, nullptr },   // 0 L1
        { wp1,  8,   8,   2, 2, 0, nullptr },   // 1 P1
        { w2,   16,  8,   3, 3, 0, nullptr },   // 2 L2
        { wp2,  16,  16,  5, 2, 0, nullptr },   // 3 P2
        { w3,   32,  16,  3, 3, 0, nullptr },   // 4 L3
        { wp3,  32,  32,  5, 2, 0, nullptr },   // 5 P3
        { mu_w, 7,   256, 1, 1, 0, nullptr },   // 6 mu
        { lv_w, 7,   256, 1, 1, 0, nullptr },   // 7 lv
        { dc_w, 256, 7,   1, 1, 0, nullptr },   // 8 dec
        { tu1w, 32,  32,  5, 2, 1, nullptr },   // 9 tu1
        { t1w,  16,  32,  3, 3, 1, nullptr },   // 10 t1
        { tu2w, 16,  16,  5, 2, 1, nullptr },   // 11 tu2
        { t2w,  8,   16,  3, 3, 1, nullptr },   // 12 t2
        { tu3w, 8,   8,   2, 2, 1, nullptr },   // 13 tu3
        { t3w,  6,   8,   3, 3, 1, nullptr },   // 14 t3
    };
    for (int i = 0; i < 15; ++i)
        pk[i].P = carve_bf16(packElems(pk[i].Cout, pk[i].Cin, pk[i].KH, pk[i].KW));
    float* mu_raw = carve_f32((size_t)B * 7);
    float* lv_raw = carve_f32((size_t)B * 7);

    // ---- pack all weights (bf16 B-fragments) ----
    for (int i = 0; i < 15; ++i) {
        int K   = pk[i].Cin * pk[i].KH * pk[i].KW;
        int nkc = (K + 31) / 32;
        int ntn = (pk[i].Cout + 15) / 16;
        int total = ntn * nkc * 512;
        pack_weights<<<(total + 255) / 256, 256, 0, stream>>>(
            pk[i].W, pk[i].P, pk[i].Cout, pk[i].Cin, pk[i].KH, pk[i].KW,
            nkc, ntn, pk[i].tflip, total);
    }

    float* outD  = (float*)d_out;                    // [B,6,50,64]
    float* outMu = outD + (size_t)B * 6 * 50 * 64;   // [B,7]
    float* outLv = outMu + (size_t)B * 7;            // [B,7]
    float* outZ  = outLv + (size_t)B * 7;            // [B,7]

    auto blocks = [](int M) { return M / 128; };     // 16 rows * 8 waves

    // ---------------- encoder ----------------
    conv_wmma<6,8,3,3, 1,1, 1,1, 1,1, 50,64, 50,64, true>
        <<<blocks(B*50*64), 256, 0, stream>>>(x, pk[0].P, b1, bufA);
    conv_wmma<8,8,2,2, 2,2, 0,0, 1,1, 50,64, 25,32, false>
        <<<blocks(B*25*32), 256, 0, stream>>>(bufA, pk[1].P, bp1, bufB);
    conv_wmma<8,16,3,3, 1,1, 1,1, 1,1, 25,32, 25,32, true>
        <<<blocks(B*25*32), 256, 0, stream>>>(bufB, pk[2].P, b2, bufA);
    conv_wmma<16,16,5,2, 5,2, 0,0, 1,1, 25,32, 5,16, false>
        <<<blocks(B*5*16), 256, 0, stream>>>(bufA, pk[3].P, bp2, bufB);
    conv_wmma<16,32,3,3, 1,1, 1,1, 1,1, 5,16, 5,16, true>
        <<<blocks(B*5*16), 256, 0, stream>>>(bufB, pk[4].P, b3, bufA);
    conv_wmma<32,32,5,2, 5,2, 0,0, 1,1, 5,16, 1,8, false>
        <<<blocks(B*1*8), 256, 0, stream>>>(bufA, pk[5].P, bp3, bufB);
    // bufB now holds f = flatten [B,256]

    // ---------------- latents ----------------
    conv_wmma<256,7,1,1, 1,1, 0,0, 1,1, 1,1, 1,1, false>
        <<<blocks(B), 256, 0, stream>>>(bufB, pk[6].P, mu_b, mu_raw);
    conv_wmma<256,7,1,1, 1,1, 0,0, 1,1, 1,1, 1,1, false>
        <<<blocks(B), 256, 0, stream>>>(bufB, pk[7].P, lv_b, lv_raw);
    vae_latent<<<(B*7 + 255) / 256, 256, 0, stream>>>(
        mu_raw, lv_raw, eps, outMu, outLv, outZ, B * 7);

    // ---------------- decoder ----------------
    conv_wmma<7,256,1,1, 1,1, 0,0, 1,1, 1,1, 1,1, true>
        <<<blocks(B), 256, 0, stream>>>(outZ, pk[8].P, dc_b, bufA);  // [B,32,1,8]
    conv_wmma<32,32,5,2, 1,1, 4,1, 2,2, 1,8, 5,16, false>           // tu1 (convT s=2,2)
        <<<blocks(B*5*16), 256, 0, stream>>>(bufA, pk[9].P, tu1b, bufB);
    conv_wmma<32,16,3,3, 1,1, 1,1, 1,1, 5,16, 5,16, true>            // t1 (convT s=1 p=1)
        <<<blocks(B*5*16), 256, 0, stream>>>(bufB, pk[10].P, t1b, bufA);
    conv_wmma<16,16,5,2, 1,1, 4,1, 5,2, 5,16, 25,32, false>          // tu2 (convT s=5,2)
        <<<blocks(B*25*32), 256, 0, stream>>>(bufA, pk[11].P, tu2b, bufB);
    conv_wmma<16,8,3,3, 1,1, 1,1, 1,1, 25,32, 25,32, true>           // t2
        <<<blocks(B*25*32), 256, 0, stream>>>(bufB, pk[12].P, t2b, bufA);
    conv_wmma<8,8,2,2, 1,1, 1,1, 2,2, 25,32, 50,64, false>           // tu3 (convT s=2,2)
        <<<blocks(B*50*64), 256, 0, stream>>>(bufA, pk[13].P, tu3b, bufB);
    conv_wmma<8,6,3,3, 1,1, 1,1, 1,1, 50,64, 50,64, true>            // t3 -> d_out
        <<<blocks(B*50*64), 256, 0, stream>>>(bufB, pk[14].P, t3b, outD);
}
